// SplineKAN_48498770706596
// MI455X (gfx1250) — compile-verified, matrix operand order checked by reading, not run on hardware
//
#include <hip/hip_runtime.h>
#include <hip/hip_bf16.h>
#include <stdint.h>

typedef __attribute__((ext_vector_type(16))) _Float16 v16h;
typedef __attribute__((ext_vector_type(8)))  float    v8f;
typedef __attribute__((ext_vector_type(4)))  uint32_t v4u;

#define IN_DIM  128
#define OUT_DIM 128
#define NUMK    5
#define KORD    3
#define BATCH   512
#define NFEAT   9                    // 1 silu + 8 spline-basis features per input scalar
#define KDIM    (IN_DIM * NFEAT)     // 1152 (== 36 * 32, exact K-chunking)
#define KCH     (KDIM / 32)          // 36 WMMA K-steps
#define KDW     (KDIM / 2)           // 576 dwords per f16 row

// ---------------------------------------------------------------------------
// Kernel 1: per-(b,i) feature evaluation  ->  F[b, i*9 + f]  (f16, K-contiguous)
// ---------------------------------------------------------------------------
__global__ void kan_features(const float* __restrict__ x,
                             const float* __restrict__ grid,
                             _Float16* __restrict__ F) {
  int idx = blockIdx.x * blockDim.x + threadIdx.x;   // 0 .. 65535
  int b = idx >> 7;
  int i = idx & 127;
  float v = x[b * IN_DIM + i];

  // silu
  float sig  = 1.0f / (1.0f + __expf(-v));
  float silu = v * sig;

  // Extended uniform knots t[0..11] from grid row (all rows identical linspace)
  float g0 = grid[0];
  float gN = grid[NUMK];
  float h  = (gN - g0) / (float)NUMK;
  float t[NUMK + 1 + 2 * KORD];                      // 12 knots
  #pragma unroll
  for (int k = 0; k < NUMK + 1 + 2 * KORD; ++k)
    t[k] = g0 + (float)(k - KORD) * h;

  // Cox–de Boor, order 0 .. 3 (half-open intervals, eps-guarded denominators)
  float B[NUMK + 2 * KORD];                          // 11 order-0 values
  #pragma unroll
  for (int k = 0; k < NUMK + 2 * KORD; ++k)
    B[k] = (v >= t[k] && v < t[k + 1]) ? 1.0f : 0.0f;
  #pragma unroll
  for (int r = 1; r <= KORD; ++r) {
    #pragma unroll
    for (int k = 0; k < NUMK + 2 * KORD - r; ++k) {
      float da    = fmaxf(t[k + r]     - t[k],     1e-14f);
      float db    = fmaxf(t[k + r + 1] - t[k + 1], 1e-14f);
      float alpha = (v - t[k]) / da;
      float beta  = (t[k + r + 1] - v) / db;
      B[k] = alpha * B[k] + beta * B[k + 1];
    }
  }

  _Float16* row = F + (size_t)b * KDIM + i * NFEAT;
  row[0] = (_Float16)silu;
  #pragma unroll
  for (int j = 0; j < NUMK + KORD; ++j)              // 8 basis funcs
    row[1 + j] = (_Float16)B[j];
}

// ---------------------------------------------------------------------------
// Kernel 2: fold mask/scales/coef into f16 weight matrix Wt[o, i*9 + f]
// ---------------------------------------------------------------------------
__global__ void kan_weights(const float* __restrict__ coef,
                            const float* __restrict__ scale_base,
                            const float* __restrict__ scale_sp,
                            const float* __restrict__ mask,
                            _Float16* __restrict__ Wt) {
  int idx = blockIdx.x * blockDim.x + threadIdx.x;   // 0 .. 16383
  int o = idx >> 7;
  int i = idx & 127;
  int s = o * IN_DIM + i;
  float mk = mask[s];
  float wb = mk * scale_base[s];
  float wsp = mk * scale_sp[s];
  _Float16* row = Wt + (size_t)o * KDIM + i * NFEAT;
  row[0] = (_Float16)wb;
  #pragma unroll
  for (int j = 0; j < NUMK + KORD; ++j)
    row[1 + j] = (_Float16)(wsp * coef[s * (NUMK + KORD) + j]);
}

// ---------------------------------------------------------------------------
// Kernel 3: y(512x128) = F(512x1152) * Wt^T(1152x128) via v_wmma_f32_16x16x32_f16
// 32 blocks x 256 threads (8 waves). Block = 16 batch rows; wave = 16 out cols.
// A panel staged in LDS (36 KB of the 320 KB WGP LDS), reused by all 8 waves.
// ---------------------------------------------------------------------------
__global__ __launch_bounds__(256)
void kan_gemm(const _Float16* __restrict__ F,
              const _Float16* __restrict__ Wt,
              float* __restrict__ out) {
  __shared__ __align__(16) uint32_t ldsA[16 * KDW];  // 16 rows x 576 dwords = 36 KB

  const int tid = threadIdx.x;
  const int bt  = blockIdx.x * 16;                   // batch-row tile base

  // Cooperative stage of the A tile (rows bt..bt+15 are contiguous in F)
  const uint32_t* Fd = (const uint32_t*)F + (size_t)bt * KDW;
  for (int tI = tid; tI < 16 * KDW; tI += 256)
    ldsA[tI] = Fd[tI];
  __syncthreads();

  const int wave = tid >> 5;
  const int lane = tid & 31;
  const int hf   = lane >> 4;                        // lane half selects K sub-block
  const int mn   = lane & 15;                        // M for A-frag, N for B/D-frag
  const int nt   = wave * 16;                        // output-column tile base

  // 16-bit A 16x32 layout: VGPR 0-3 hold K = hf*8 + 0..7 (2 halves/VGPR),
  //                        VGPR 4-7 hold K = 16 + hf*8 + 0..7
  // => two 16B loads at dword offsets hf*4 and 8 + hf*4 within each K-chunk.
  const uint32_t* arow = ldsA + mn * KDW + hf * 4;
  const uint32_t* brow = (const uint32_t*)Wt + (size_t)(nt + mn) * KDW + hf * 4;

  union AB { v4u u[2]; v16h h; };
  v8f acc = {};

  for (int kc = 0; kc < KCH; ++kc) {
    AB a, b;
    a.u[0] = *(const v4u*)(arow + kc * 16);          // ds_load_b128
    a.u[1] = *(const v4u*)(arow + kc * 16 + 8);
    b.u[0] = *(const v4u*)(brow + kc * 16);          // global_load_b128
    b.u[1] = *(const v4u*)(brow + kc * 16 + 8);
    __builtin_prefetch(brow + (kc + 1) * 16, 0, 1);  // global_prefetch_b8 (speculative)
    // D = A * B + C, f32 accumulate
    acc = __builtin_amdgcn_wmma_f32_16x16x32_f16(false, a.h, false, b.h,
                                                 (short)0, acc, false, false);
  }

  // 16x16 f32 D layout: VGPR v <-> M = v + 8*hf, N = mn (per lane half)
  float* orow = out + (size_t)(bt + hf * 8) * OUT_DIM + nt + mn;
  #pragma unroll
  for (int v = 0; v < 8; ++v)
    orow[(size_t)v * OUT_DIM] = acc[v];
}

// ---------------------------------------------------------------------------
// Launcher. Inputs (setup_inputs order): x, grid, coef, scale_base, scale_sp, mask
// Workspace: F (512*1152 f16 = 1.125 MB) then Wt (128*1152 f16 = 288 KB).
// ---------------------------------------------------------------------------
extern "C" void kernel_launch(void* const* d_in, const int* in_sizes, int n_in,
                              void* d_out, int out_size, void* d_ws, size_t ws_size,
                              hipStream_t stream) {
  const float* x    = (const float*)d_in[0];
  const float* grid = (const float*)d_in[1];
  const float* coef = (const float*)d_in[2];
  const float* sb   = (const float*)d_in[3];
  const float* ssp  = (const float*)d_in[4];
  const float* mk   = (const float*)d_in[5];
  float* out = (float*)d_out;

  _Float16* F  = (_Float16*)d_ws;
  _Float16* Wt = (_Float16*)((char*)d_ws + (size_t)BATCH * KDIM * sizeof(_Float16));

  kan_features<<<(BATCH * IN_DIM) / 256, 256, 0, stream>>>(x, grid, F);
  kan_weights<<<(OUT_DIM * IN_DIM) / 128, 128, 0, stream>>>(coef, sb, ssp, mk, Wt);
  kan_gemm<<<BATCH / 16, 256, 0, stream>>>(F, Wt, out);
}
